// MipRayMarcher2_44281112822421
// MI455X (gfx1250) — matrix operand order repaired and checked by reading, not compiled, not run on hardware
//
#include <hip/hip_runtime.h>
#include <hip/hip_bf16.h>
#include <math.h>

// CDNA5 / gfx1250: wave32, WMMA 16x16x32 f16->f32
typedef __attribute__((ext_vector_type(16))) _Float16 v16h;
typedef __attribute__((ext_vector_type(8)))  float    v8f;
typedef __attribute__((ext_vector_type(4)))  float    v4f;

// Problem constants (reference: B=4, R=16384, S=64, C=32)
#define S_SAMPLES 64
#define S_INT     63
#define C_CH      32
#define RAYS_PER_BLOCK 16   // matches WMMA N dimension

// ---- ordered-uint mapping for float min/max atomics ----
__device__ __forceinline__ unsigned f2key(float f) {
  unsigned u = __float_as_uint(f);
  return (u & 0x80000000u) ? ~u : (u | 0x80000000u);
}
__device__ __forceinline__ float key2f(unsigned k) {
  unsigned u = (k & 0x80000000u) ? (k ^ 0x80000000u) : ~k;
  return __uint_as_float(u);
}

__global__ void init_ws(unsigned* ws) {
  ws[0] = 0xFFFFFFFFu;  // min-key init (maps to +inf)
  ws[1] = 0x00000000u;  // max-key init (maps to -inf)
}

// Global depth min/max: grid-stride + wave reduce + one atomic per wave.
__global__ void __launch_bounds__(256) depth_minmax(const float* __restrict__ depths,
                                                    int n, unsigned* __restrict__ ws) {
  unsigned kmin = 0xFFFFFFFFu, kmax = 0u;
  for (int i = blockIdx.x * blockDim.x + threadIdx.x; i < n;
       i += gridDim.x * blockDim.x) {
    unsigned k = f2key(depths[i]);
    kmin = min(kmin, k);
    kmax = max(kmax, k);
  }
#pragma unroll
  for (int off = 16; off >= 1; off >>= 1) {  // wave32 butterfly
    kmin = min(kmin, (unsigned)__shfl_xor((int)kmin, off));
    kmax = max(kmax, (unsigned)__shfl_xor((int)kmax, off));
  }
  if ((threadIdx.x & 31) == 0) {
    atomicMin(&ws[0], kmin);
    atomicMax(&ws[1], kmax);
  }
}

// Main kernel: one block = 16 rays. 256 threads = 8 waves.
//  A) per-interval scalar pipeline -> LDS (alpha, x=-log t, depth_mid, grad_mid)
//  B) wave 0: exclusive cumsum of x over 64 rows x 16 rays via WMMA with a
//     lower-triangular ones matrix (L is exact in f16); weights = alpha*exp(-y)
//  C) per-sample color weights cw[s] = 0.5*(w[s-1]+w[s]); per-ray reductions
//  D) all 8 waves stream colors (non-temporal B128) and dot with cw
__global__ void __launch_bounds__(256) raymarch_kernel(
    const float* __restrict__ colors, const float* __restrict__ densities,
    const float* __restrict__ depths, const float* __restrict__ grads,
    float* __restrict__ out_rgb, float* __restrict__ out_depth,
    float* __restrict__ out_w, float* __restrict__ out_grad,
    const unsigned* __restrict__ ws) {
  __shared__ float xl[RAYS_PER_BLOCK][S_SAMPLES];      // -log(exp(-dd)+1e-10)
  __shared__ float al[RAYS_PER_BLOCK][S_SAMPLES];      // alpha
  __shared__ float dm[RAYS_PER_BLOCK][S_SAMPLES];      // depths_mid
  __shared__ float gm[RAYS_PER_BLOCK][S_SAMPLES][3];   // grads_mid
  __shared__ float wt[RAYS_PER_BLOCK][S_SAMPLES];      // weights
  __shared__ float cw[RAYS_PER_BLOCK][S_SAMPLES];      // per-sample color weight

  const int tid  = threadIdx.x;
  const int lane = tid & 31;
  const int wid  = tid >> 5;
  const int g0   = blockIdx.x * RAYS_PER_BLOCK;

  // ---------------- Phase A ----------------
  {
    const int r    = tid >> 4;     // 0..15 ray within block
    const int i0   = tid & 15;
    const int base = (g0 + r) * S_SAMPLES;
#pragma unroll
    for (int k = 0; k < 4; ++k) {
      const int i = i0 + 16 * k;   // 0..63
      if (i < S_INT) {
        float d0 = densities[base + i], d1 = densities[base + i + 1];
        float z0 = depths[base + i],    z1 = depths[base + i + 1];
        float sm = 0.5f * (d0 + d1) - 1.0f;
        float sp = fmaxf(sm, 0.0f) + log1pf(expf(-fabsf(sm)));  // softplus
        float dd = sp * (z1 - z0);
        float e  = expf(-dd);
        al[r][i] = 1.0f - e;
        xl[r][i] = -logf(e + 1e-10f);
        dm[r][i] = 0.5f * (z0 + z1);
#pragma unroll
        for (int c = 0; c < 3; ++c)
          gm[r][i][c] = 0.5f * (grads[(base + i) * 3 + c] +
                                grads[(base + i + 1) * 3 + c]);
      } else {  // pad row 63
        xl[r][63] = 0.0f; al[r][63] = 0.0f; dm[r][63] = 0.0f;
        gm[r][63][0] = gm[r][63][1] = gm[r][63][2] = 0.0f;
      }
    }
  }
  __syncthreads();

  // ---------------- Phase B : WMMA triangular-matmul scan (wave 0) -------
  if (wid == 0) {
    const int n  = lane & 15;   // A-row M, B/D column N (ray)
    const int hi = lane >> 4;   // lane half selects K/M sub-ranges
#pragma unroll
    for (int m = 0; m < 4; ++m) {        // 16-row output tiles: rows 16m..16m+15
      v8f acc = {};
      const int ig = 16 * m + n;         // global row index held by this lane (A)
#pragma unroll
      for (int kt = 0; kt <= (m >> 1); ++kt) {   // K tiles of 32
        v16h a, b;
#pragma unroll
        for (int h = 0; h < 16; ++h) {
          // A 16x32 f16 layout: lanes0-15 K in {0..7,16..23}; lanes16-31 +8
          const int Ka = (hi == 0) ? (h < 8 ? h : h + 8)
                                   : (h < 8 ? h + 8 : h + 16);
          const int ja = 32 * kt + Ka;
          a[h] = (ja < ig) ? (_Float16)1.0f : (_Float16)0.0f;  // strict lower tri
          // B 32x16 f16 layout: same K order as A (lanes0-15 K in {0..7,16..23}; lanes16-31 +8)
          const int jb = 32 * kt + Ka;
          b[h] = (_Float16)xl[n][jb];
        }
        acc = __builtin_amdgcn_wmma_f32_16x16x32_f16(
            false, a, false, b, (short)0, acc, false, false);
      }
      // D 16x16 f32 layout: VGPR g -> (M = g + 8*hi, N = lane%16)
#pragma unroll
      for (int g = 0; g < 8; ++g) {
        const int i = 16 * m + g + 8 * hi;
        if (i < S_INT) {
          const float w = al[n][i] * expf(-acc[g]);  // alpha * excl. transmittance
          wt[n][i] = w;
          out_w[(size_t)(g0 + n) * S_INT + i] = w;
        }
      }
    }
  }
  __syncthreads();

  // ---------------- Phase C ----------------
  {
    const int r = tid >> 4, s0 = tid & 15;
#pragma unroll
    for (int k = 0; k < 4; ++k) {
      const int s = s0 + 16 * k;
      const float wp = (s > 0)     ? wt[r][s - 1] : 0.0f;
      const float wc = (s < S_INT) ? wt[r][s]     : 0.0f;
      cw[r][s] = 0.5f * (wp + wc);   // folds midpoint averaging of colors
    }
  }
  if (tid < RAYS_PER_BLOCK) {
    const int r = tid;
    float wsum = 0.f, dsum = 0.f, gs0 = 0.f, gs1 = 0.f, gs2 = 0.f;
    for (int i = 0; i < S_INT; ++i) {
      const float w = wt[r][i];
      wsum += w;
      dsum += w * dm[r][i];
      gs0 += w * gm[r][i][0];
      gs1 += w * gm[r][i][1];
      gs2 += w * gm[r][i][2];
    }
    const float dmin = key2f(ws[0]), dmax = key2f(ws[1]);
    float cd = dsum / wsum;
    if (isnan(cd)) cd = __builtin_inff();
    cd = fminf(fmaxf(cd, dmin), dmax);
    const int rg = g0 + r;
    out_depth[rg] = cd;
    out_grad[rg * 3 + 0] = gs0 + 1.0f - wsum;
    out_grad[rg * 3 + 1] = gs1 + 1.0f - wsum;
    out_grad[rg * 3 + 2] = gs2 + 1.0f - wsum;
  }
  __syncthreads();

  // ---------------- Phase D : stream colors (NT, B128) -------------------
  // lane -> (channel quad cq, sample-row offset soff); wave covers 4 sample
  // rows x 32 channels per iteration.
  const int cq = lane & 7, soff = lane >> 3;
#pragma unroll
  for (int rr = 0; rr < 2; ++rr) {               // 2 rays per wave
    const int r  = 2 * wid + rr;
    const int rg = g0 + r;
    const v4f* cp = (const v4f*)(colors + (size_t)rg * (S_SAMPLES * C_CH));
    v4f acc = {0.f, 0.f, 0.f, 0.f};
#pragma unroll
    for (int k = 0; k < 16; ++k) {
      const int s = 4 * k + soff;
      const v4f col = __builtin_nontemporal_load(cp + s * 8 + cq);
      acc += cw[r][s] * col;
    }
    // reduce across the 4 lanes sharing a channel quad (xor 8, 16)
#pragma unroll
    for (int off = 8; off <= 16; off <<= 1) {
      acc.x += __shfl_xor(acc.x, off);
      acc.y += __shfl_xor(acc.y, off);
      acc.z += __shfl_xor(acc.z, off);
      acc.w += __shfl_xor(acc.w, off);
    }
    if (lane < 8) {
      v4f outv = acc * 2.0f - 1.0f;
      *(v4f*)(out_rgb + (size_t)rg * C_CH + cq * 4) = outv;
    }
  }
}

extern "C" void kernel_launch(void* const* d_in, const int* in_sizes, int n_in,
                              void* d_out, int out_size, void* d_ws, size_t ws_size,
                              hipStream_t stream) {
  const float* colors    = (const float*)d_in[0];
  const float* densities = (const float*)d_in[1];
  const float* depths    = (const float*)d_in[2];
  const float* grads     = (const float*)d_in[3];

  const int nrays   = in_sizes[1] / S_SAMPLES;  // B*R = 65536
  const int ndepths = in_sizes[2];

  unsigned* ws = (unsigned*)d_ws;
  float* out       = (float*)d_out;
  float* out_rgb   = out;                                   // [B,R,32]
  float* out_depth = out_rgb + (size_t)nrays * C_CH;        // [B,R,1]
  float* out_w     = out_depth + nrays;                     // [B,R,63,1]
  float* out_grad  = out_w + (size_t)nrays * S_INT;         // [B,R,3]

  init_ws<<<1, 1, 0, stream>>>(ws);
  depth_minmax<<<512, 256, 0, stream>>>(depths, ndepths, ws);
  raymarch_kernel<<<nrays / RAYS_PER_BLOCK, 256, 0, stream>>>(
      colors, densities, depths, grads, out_rgb, out_depth, out_w, out_grad, ws);
}
